// MultiHeadAttention_13855564497151
// MI455X (gfx1250) — matrix operand order repaired and checked by reading, not compiled, not run on hardware
//
#include <hip/hip_runtime.h>
#include <hip/hip_bf16.h>
#include <math.h>
#include <stdint.h>

#define DEV static __device__ __forceinline__

typedef __attribute__((ext_vector_type(16))) __bf16 v16bf;
typedef __attribute__((ext_vector_type(8)))  __bf16 v8bf;
typedef __attribute__((ext_vector_type(8)))  float  v8f;
typedef __attribute__((ext_vector_type(4)))  unsigned int u32x4;
typedef __attribute__((ext_vector_type(8)))  int i32x8;
typedef __attribute__((ext_vector_type(4)))  int i32x4;

constexpr int Bd  = 2;     // batch
constexpr int Sd  = 2048;  // sequence
constexpr int CIN = 512;   // model dim
constexpr int Hh  = 8;     // heads
constexpr int Ch  = 64;    // head dim
constexpr int Dd  = 512;   // H*C

// ---- CDNA5 TDM detection (device pass; host pass parses the asm fallback) ----
#if defined(__has_builtin)
#  if __has_builtin(__builtin_amdgcn_tensor_load_to_lds) && \
      __has_builtin(__builtin_amdgcn_s_wait_tensorcnt)
#    define HAVE_TDM 1
#  endif
#endif

DEV v16bf cat16(v8bf lo, v8bf hi) {
  return __builtin_shufflevector(lo, hi, 0,1,2,3,4,5,6,7,8,9,10,11,12,13,14,15);
}
DEV v8bf ld8(const __bf16* p) { return *(const v8bf*)p; }

// A-fragment (16xK=32, bf16): lane<16 -> row=lane, K={0..7,16..23};
// lane>=16 -> row=lane-16, K={8..15,24..31}. base points at (row0,k0), ld = row stride.
DEV v16bf loadA(const __bf16* base, int ld, int lane) {
  const __bf16* p = base + (size_t)(lane & 15) * ld + 8 * (lane >> 4);
  return cat16(ld8(p), ld8(p + 16));
}
// B-fragment (K=32 x 16, bf16): lane<16 -> col=lane, K={0..7,16..23};
// lane>=16 -> col=lane-16, K={8..15,24..31}. base at (k0,n0) of [N][K]-major array.
DEV v16bf loadB(const __bf16* base, int ld, int lane) {
  const __bf16* p = base + (size_t)(lane & 15) * ld + 8 * (lane >> 4);
  return cat16(ld8(p), ld8(p + 16));
}

DEV v8f wmma_bf16(v16bf a, v16bf b, v8f c) {
  return __builtin_amdgcn_wmma_f32_16x16x32_bf16(false, a, false, b, (short)0, c,
                                                 false, false);
}

// ---- TDM: DMA a 2-D bf16 tile [rows x cols] from global to LDS -------------
#if defined(HAVE_TDM)
DEV void tdm_load_2d_bf16(void* lds, const void* gaddr, int rows, int cols,
                          int tensor_rows, int stride_elems) {
  const unsigned long long ga = (unsigned long long)(uintptr_t)gaddr;
  u32x4 g0;
  g0[0] = 1u;                                            // count=1, user D#
  g0[1] = (unsigned)(uintptr_t)lds;                      // lds_addr (low 32 bits)
  g0[2] = (unsigned)(ga & 0xffffffffu);                  // global_addr[31:0]
  g0[3] = (unsigned)((ga >> 32) & 0x01ffffffu) | (2u << 30);  // addr[56:32]|type=2
  i32x8 g1;
  g1[0] = (1 << 16);                                     // wg_mask=0, data_size=2B
  g1[1] = (int)((unsigned)cols << 16);                   // tensor_dim0[15:0] @63:48
  g1[2] = (int)((unsigned)tensor_rows << 16);            // tensor_dim1[15:0] @95:80
  g1[3] = (int)((unsigned)cols << 16);                   // tile_dim0 @127:112
  g1[4] = rows;                                          // tile_dim1 @143:128
  g1[5] = stride_elems;                                  // tensor_dim0_stride[31:0]
  g1[6] = 0;                                             // stride hi / dim1_stride
  g1[7] = 0;
  const i32x4 z = {0, 0, 0, 0};
#if (__clang_major__ <= 22)
  __builtin_amdgcn_tensor_load_to_lds(g0, g1, z, z, 0);            // ROCm 7.2 5-arg
#else
  const i32x8 z8 = {0, 0, 0, 0, 0, 0, 0, 0};
  __builtin_amdgcn_tensor_load_to_lds(g0, g1, z, z, z8, 0);        // clang-23 6-arg
#endif
}
DEV void tdm_wait() { __builtin_amdgcn_s_wait_tensorcnt(0); }
#else
// ---- fallback: async global->LDS 16-byte copies (ASYNCcnt path) ------------
DEV void async_cp16(__bf16* lds, const __bf16* g) {
  asm volatile("global_load_async_to_lds_b128 %0, %1, off"
               :: "v"((unsigned)(uintptr_t)lds), "v"(g) : "memory");
}
DEV void async_wait() { asm volatile("s_wait_asynccnt 0x0" ::: "memory"); }
#endif

// ---------------------------------------------------------------- converters
__global__ void cvt_f32_bf16(const float* __restrict__ s, __bf16* __restrict__ d, int n) {
  int i = (blockIdx.x * blockDim.x + threadIdx.x) * 8;
  if (i >= n) return;
  float4 a = *(const float4*)(s + i);
  float4 b = *(const float4*)(s + i + 4);
  v8bf r;
  r[0] = (__bf16)a.x; r[1] = (__bf16)a.y; r[2] = (__bf16)a.z; r[3] = (__bf16)a.w;
  r[4] = (__bf16)b.x; r[5] = (__bf16)b.y; r[6] = (__bf16)b.z; r[7] = (__bf16)b.w;
  *(v8bf*)(d + i) = r;
}

// ---------------------------------------------------------------- generic GEMM
// Out[M,512] = A[M,512](bf16) @ W[512,512]^T(bf16), epilogue by mode:
//  0: *1/sqrt(64), bf16 head-split [B,H,S,64]   (Q)
//  1,2: bf16 head-split                          (K,V)
//  3: sigmoid(v + bg[col]) fp32 [M,512]          (gate)
//  4: v + bo[col] fp32 [M,512]                   (final output)
__global__ __launch_bounds__(128) void gemm_bf16(
    const __bf16* __restrict__ A, const __bf16* __restrict__ W,
    int mode, void* __restrict__ outp, const float* __restrict__ bvec)
{
  const int lane = threadIdx.x & 31, w = threadIdx.x >> 5;
  const int r0 = blockIdx.x * 64 + w * 16;
  const int c0g = blockIdx.y * 64;
  const int half = lane >> 4, nl = lane & 15;

  v8f acc[4] = {};
  for (int k0 = 0; k0 < CIN; k0 += 32) {
    v16bf a = loadA(A + (size_t)r0 * CIN + k0, CIN, lane);
#pragma unroll
    for (int nt = 0; nt < 4; ++nt) {
      v16bf b = loadB(W + (size_t)(c0g + nt * 16) * CIN + k0, CIN, lane);
      acc[nt] = wmma_bf16(a, b, acc[nt]);
    }
  }
#pragma unroll
  for (int nt = 0; nt < 4; ++nt) {
    const int col = c0g + nt * 16 + nl;
#pragma unroll
    for (int vg = 0; vg < 8; ++vg) {
      const int row = r0 + vg + 8 * half;
      float v = acc[nt][vg];
      if (mode == 0) v *= 0.125f;            // 1/sqrt(head_dim=64)
      if (mode <= 2) {
        const int b = row >> 11, s = row & (Sd - 1);
        const int h = col >> 6,  c = col & (Ch - 1);
        ((__bf16*)outp)[(((size_t)(b * Hh + h)) * Sd + s) * Ch + c] = (__bf16)v;
      } else if (mode == 3) {
        v += bvec[col];
        ((float*)outp)[(size_t)row * Dd + col] = 1.0f / (1.0f + __expf(-v));
      } else {
        ((float*)outp)[(size_t)row * CIN + col] = v + bvec[col];
      }
    }
  }
}

// ---------------------------------------------------------------- fused attention
// grid: (S/64, B*H), block 128 (4 waves). Wave w owns q-rows [bx*64+w*16, +16).
// K/V tiles staged by the Tensor Data Mover (or async-to-LDS asm); flash-style
// online softmax in fp32; P reshaped to an A-fragment via per-wave LDS.
__global__ __launch_bounds__(128) void attn_fused(
    const __bf16* __restrict__ Q, const __bf16* __restrict__ K,
    const __bf16* __restrict__ V, const float* __restrict__ bias,
    const float* __restrict__ G, __bf16* __restrict__ O)
{
  __shared__ __bf16 Kl[32 * 64];     // [krow][ch]
  __shared__ __bf16 Vr[32 * 64];     // [krow][ch] raw (DMA target)
  __shared__ __bf16 Vt[64 * 32];     // [ch][krow] transposed
  __shared__ __bf16 Pl[4][16 * 32];  // per-wave P tile [qrow][kcol]

  const int tid = threadIdx.x, lane = tid & 31, w = tid >> 5;
  const int bh = blockIdx.y;                 // b*H + h
  const int b = bh >> 3, h = bh & 7;
  const int q0 = blockIdx.x * 64 + w * 16;
  const int half = lane >> 4, nl = lane & 15;

  const __bf16* qbase = Q + ((size_t)bh * Sd + q0) * Ch;
  const v16bf fq0 = loadA(qbase + 0, Ch, lane);
  const v16bf fq1 = loadA(qbase + 32, Ch, lane);
  const float* brow = bias + ((size_t)bh * Sd + q0) * Sd;

  float mrun[8], lrun[8];
  v8f accO[4] = {};
#pragma unroll
  for (int i = 0; i < 8; ++i) { mrun[i] = -1e30f; lrun[i] = 0.0f; }

  for (int kb = 0; kb < Sd; kb += 32) {
    const __bf16* kg = K + ((size_t)bh * Sd + kb) * Ch;
    const __bf16* vg = V + ((size_t)bh * Sd + kb) * Ch;

    // ---- stage K and raw V tiles into LDS (TDM preferred, else async asm) ----
#if defined(HAVE_TDM)
    if (w == 0) {                 // one wave drives the DMA engine
      tdm_load_2d_bf16(Kl, kg, 32, Ch, Sd, Ch);
      tdm_load_2d_bf16(Vr, vg, 32, Ch, Sd, Ch);
      tdm_wait();
    }
#else
    {
      const int r = tid >> 2, c = (tid & 3) << 4;      // r:0..31, c:{0,16,32,48}
      async_cp16(&Kl[r * 64 + c],     kg + (size_t)r * 64 + c);
      async_cp16(&Kl[r * 64 + c + 8], kg + (size_t)r * 64 + c + 8);
      async_cp16(&Vr[r * 64 + c],     vg + (size_t)r * 64 + c);
      async_cp16(&Vr[r * 64 + c + 8], vg + (size_t)r * 64 + c + 8);
      async_wait();
    }
#endif
    __syncthreads();

    // ---- transpose V inside LDS: Vr[r][c] -> Vt[c][r] ----
    {
      const int r = tid >> 2, c = (tid & 3) << 4;
#pragma unroll
      for (int i = 0; i < 16; ++i) Vt[(c + i) * 32 + r] = Vr[r * 64 + c + i];
    }
    __syncthreads();

    if (kb + 32 < Sd)  // gfx1250 global_prefetch_b8 of next bias tile
      __builtin_prefetch(brow + (size_t)(half * 8) * Sd + kb + 32 + nl, 0, 1);

    // ---- scores S = Q(16x64) @ K^T(64x32) ----
    v8f sc[2] = {};
#pragma unroll
    for (int nt = 0; nt < 2; ++nt) {
      sc[nt] = wmma_bf16(fq0, loadB(&Kl[(nt * 16) * 64 + 0], 64, lane), sc[nt]);
      sc[nt] = wmma_bf16(fq1, loadB(&Kl[(nt * 16) * 64 + 32], 64, lane), sc[nt]);
    }
    // add pair bias (fp32, read exactly once per element)
#pragma unroll
    for (int nt = 0; nt < 2; ++nt)
#pragma unroll
      for (int vg = 0; vg < 8; ++vg)
        sc[nt][vg] += brow[(size_t)(vg + 8 * half) * Sd + kb + nt * 16 + nl];

    // ---- online softmax (row reductions across 16-lane column group) ----
    __bf16* pw = Pl[w];
#pragma unroll
    for (int vg = 0; vg < 8; ++vg) {
      float s0 = sc[0][vg], s1 = sc[1][vg];
      float mx = fmaxf(s0, s1);
#pragma unroll
      for (int msk = 8; msk >= 1; msk >>= 1) mx = fmaxf(mx, __shfl_xor(mx, msk, 32));
      const float mnew = fmaxf(mrun[vg], mx);
      const float p0 = __expf(s0 - mnew), p1 = __expf(s1 - mnew);
      float ps = p0 + p1;
#pragma unroll
      for (int msk = 8; msk >= 1; msk >>= 1) ps += __shfl_xor(ps, msk, 32);
      const float alpha = __expf(mrun[vg] - mnew);
      lrun[vg] = lrun[vg] * alpha + ps;
      mrun[vg] = mnew;
#pragma unroll
      for (int t = 0; t < 4; ++t) accO[t][vg] *= alpha;
      const int m = vg + 8 * half;
      pw[m * 32 + nl]      = (__bf16)p0;
      pw[m * 32 + 16 + nl] = (__bf16)p1;
    }

    // reshape P (C/D layout) -> A-fragment through per-wave LDS (DS in-order)
    const __bf16* pp = pw + (size_t)nl * 32 + 8 * half;
    const v16bf pa = cat16(ld8(pp), ld8(pp + 16));

    // ---- O += P(16x32) @ V(32x64) ----
#pragma unroll
    for (int t = 0; t < 4; ++t)
      accO[t] = wmma_bf16(pa, loadB(&Vt[(t * 16) * 32], 32, lane), accO[t]);

    __syncthreads();
  }

  // ---- normalize, gate, store merged-head bf16 ----
#pragma unroll
  for (int vg = 0; vg < 8; ++vg) {
    const int m = vg + 8 * half;
    const float inv = 1.0f / lrun[vg];
    const size_t orow = ((size_t)b * Sd + q0 + m) * Dd + h * Ch;
#pragma unroll
    for (int t = 0; t < 4; ++t) {
      const int c = t * 16 + nl;
      const float o = accO[t][vg] * inv * G[orow + c];
      O[orow + c] = (__bf16)o;
    }
  }
}

// ---------------------------------------------------------------- launcher
extern "C" void kernel_launch(void* const* d_in, const int* in_sizes, int n_in,
                              void* d_out, int out_size, void* d_ws, size_t ws_size,
                              hipStream_t stream) {
  const float* x    = (const float*)d_in[0];
  const float* bias = (const float*)d_in[1];
  const float* Wq   = (const float*)d_in[2];
  const float* Wk   = (const float*)d_in[3];
  const float* Wv   = (const float*)d_in[4];
  const float* Wo   = (const float*)d_in[5];
  const float* bo   = (const float*)d_in[6];
  const float* Wg   = (const float*)d_in[7];
  const float* bg   = (const float*)d_in[8];

  const size_t M = (size_t)Bd * Sd;  // 4096
  char* ws = (char*)d_ws;
  __bf16* xb   = (__bf16*)ws; ws += M * CIN * 2;
  __bf16* wqb  = (__bf16*)ws; ws += (size_t)Dd * CIN * 2;
  __bf16* wkb  = (__bf16*)ws; ws += (size_t)Dd * CIN * 2;
  __bf16* wvb  = (__bf16*)ws; ws += (size_t)Dd * CIN * 2;
  __bf16* wgb  = (__bf16*)ws; ws += (size_t)Dd * CIN * 2;
  __bf16* wob  = (__bf16*)ws; ws += (size_t)CIN * Dd * 2;
  __bf16* qb   = (__bf16*)ws; ws += M * Dd * 2;
  __bf16* kbuf = (__bf16*)ws; ws += M * Dd * 2;
  __bf16* vbuf = (__bf16*)ws; ws += M * Dd * 2;
  float*  gbuf = (float*)ws;  ws += M * Dd * 4;
  __bf16* obuf = (__bf16*)ws; ws += M * Dd * 2;

  const int nx = (int)(M * CIN), nw = Dd * CIN;
  cvt_f32_bf16<<<(nx / 8 + 255) / 256, 256, 0, stream>>>(x, xb, nx);
  cvt_f32_bf16<<<(nw / 8 + 255) / 256, 256, 0, stream>>>(Wq, wqb, nw);
  cvt_f32_bf16<<<(nw / 8 + 255) / 256, 256, 0, stream>>>(Wk, wkb, nw);
  cvt_f32_bf16<<<(nw / 8 + 255) / 256, 256, 0, stream>>>(Wv, wvb, nw);
  cvt_f32_bf16<<<(nw / 8 + 255) / 256, 256, 0, stream>>>(Wg, wgb, nw);
  cvt_f32_bf16<<<(nw / 8 + 255) / 256, 256, 0, stream>>>(Wo, wob, nw);

  const dim3 ggrid(M / 64, Dd / 64);  // (64, 8)
  gemm_bf16<<<ggrid, 128, 0, stream>>>(xb, wqb, 0, qb, nullptr);
  gemm_bf16<<<ggrid, 128, 0, stream>>>(xb, wkb, 1, kbuf, nullptr);
  gemm_bf16<<<ggrid, 128, 0, stream>>>(xb, wvb, 2, vbuf, nullptr);
  gemm_bf16<<<ggrid, 128, 0, stream>>>(xb, wgb, 3, gbuf, bg);

  attn_fused<<<dim3(Sd / 64, Bd * Hh), 128, 0, stream>>>(qb, kbuf, vbuf, bias,
                                                         gbuf, obuf);

  gemm_bf16<<<ggrid, 128, 0, stream>>>(obuf, wob, 4, d_out, bo);
}
